// MultiHeadAttention_72241349919331
// MI455X (gfx1250) — compile-verified, matrix operand order checked by reading, not run on hardware
//
#include <hip/hip_runtime.h>
#include <hip/hip_bf16.h>
#include <math.h>

// ---------------------------------------------------------------------------
// MHA forward for MI455X (gfx1250, wave32, WMMA bf16 path)
// B=1, S=4096, E=768, H=12, Dh=64
// ---------------------------------------------------------------------------

typedef __bf16 bf16_t;
typedef __attribute__((ext_vector_type(16))) __bf16 v16bf;
typedef __attribute__((ext_vector_type(8)))  __bf16 v8bf;
typedef __attribute__((ext_vector_type(8)))  float  v8f;
typedef __attribute__((ext_vector_type(4)))  float  v4f;

#define S_LEN 4096
#define EMB   768
#define NH    12
#define DH    64

static __device__ __forceinline__ v16bf load_frag16(const bf16_t* p) {
    v8bf lo = *(const v8bf*)(p);
    v8bf hi = *(const v8bf*)(p + 16);
    return __builtin_shufflevector(lo, hi, 0,1,2,3,4,5,6,7,8,9,10,11,12,13,14,15);
}

static __device__ __forceinline__ v8f wmma_bf16(v16bf a, v16bf b, v8f c) {
    return __builtin_amdgcn_wmma_f32_16x16x32_bf16(false, a, false, b, (short)0, c,
                                                   false, false);
}

// ---------------------------------------------------------------------------
// Kernel 1: fp32 -> bf16 conversion (4 elements / thread).
// ---------------------------------------------------------------------------
__global__ void convert_acts_kernel(const float* __restrict__ q,
                                    const float* __restrict__ k,
                                    const float* __restrict__ v,
                                    bf16_t* __restrict__ X) {
    const int A = S_LEN * EMB;
    int i = (blockIdx.x * blockDim.x + threadIdx.x) * 4;
    const float* src;
    int off;
    if (i < A)            { src = q; off = i; }
    else if (i < 2 * A)   { src = k; off = i - A; }
    else                  { src = v; off = i - 2 * A; }
    v4f f = *(const v4f*)(src + off);
    bf16_t* dst = X + i;
    dst[0] = (bf16_t)f.x; dst[1] = (bf16_t)f.y;
    dst[2] = (bf16_t)f.z; dst[3] = (bf16_t)f.w;
}

// ---------------------------------------------------------------------------
// Kernel 2: transpose+convert the four weight matrices: Wt[n][k] = (bf16)W[k][n]
// grid: (EMB/256, EMB, 4)
// ---------------------------------------------------------------------------
__global__ void transpose_w_kernel(const float* __restrict__ w0,
                                   const float* __restrict__ w1,
                                   const float* __restrict__ w2,
                                   const float* __restrict__ w3,
                                   bf16_t* __restrict__ Wt) {
    const int z = blockIdx.z;
    const float* W = (z == 0) ? w0 : (z == 1) ? w1 : (z == 2) ? w2 : w3;
    bf16_t* dst = Wt + (size_t)z * EMB * EMB;
    const int n = blockIdx.x * blockDim.x + threadIdx.x;   // coalesced read dim
    const int kk = blockIdx.y;
    dst[(size_t)n * EMB + kk] = (bf16_t)W[(size_t)kk * EMB + n];
}

// ---------------------------------------------------------------------------
// Kernel 3: QKV projection GEMM.  X[4096x768](bf16) @ W[768x768] + b.
// One wave computes a 32x64 output tile: 2 A-frags x 4 B-frags = 8 WMMAs per
// 32-deep k-step, explicitly double-buffered so next-step fragments are in
// flight during the current WMMAs.
// z = 0: Q -> Qb[h][s][64] ; z = 1: K -> Kb[h][s][64] ; z = 2: V -> Vt[h][d][s]
// grid: (192, 1, 3), block 256 (8 waves)
// ---------------------------------------------------------------------------
__global__ void gemm_qkv_kernel(const bf16_t* __restrict__ Xbase,
                                const bf16_t* __restrict__ Wtbase,
                                const float* __restrict__ bq,
                                const float* __restrict__ bk,
                                const float* __restrict__ bv,
                                bf16_t* __restrict__ Qb,
                                bf16_t* __restrict__ Kb,
                                bf16_t* __restrict__ Vt) {
    const int z    = blockIdx.z;
    const int lane = threadIdx.x & 31;
    const int wave = threadIdx.x >> 5;
    const int wgid = blockIdx.x * (blockDim.x >> 5) + wave;   // 0..1535
    const int mt   = wgid / (EMB / 64);                        // 0..127
    const int nt   = wgid % (EMB / 64);                        // 0..11
    const int m0   = mt * 32;
    const int n0   = nt * 64;
    const int kh   = lane >> 4;
    const int l15  = lane & 15;

    const bf16_t* X  = Xbase  + (size_t)z * (S_LEN * EMB);
    const bf16_t* Wt = Wtbase + (size_t)z * (EMB * EMB);
    const float* bias = (z == 0) ? bq : (z == 1) ? bk : bv;

    const bf16_t* ap0 = X  + (size_t)(m0 + l15) * EMB + kh * 8;
    const bf16_t* ap1 = X  + (size_t)(m0 + 16 + l15) * EMB + kh * 8;
    const bf16_t* bp  = Wt + (size_t)(n0 + l15) * EMB + kh * 8;

    v8f acc[8] = {};

    // prologue: fragments for k=0
    v16bf a0 = load_frag16(ap0);
    v16bf a1 = load_frag16(ap1);
    v16bf b0 = load_frag16(bp);
    v16bf b1 = load_frag16(bp + 16 * EMB);
    v16bf b2 = load_frag16(bp + 32 * EMB);
    v16bf b3 = load_frag16(bp + 48 * EMB);

    for (int k0 = 0; k0 < EMB; k0 += 32) {
        int kn = k0 + 32;
        if (kn >= EMB) kn = 0;                 // last iter: dummy (valid) refill
        __builtin_prefetch(ap0 + kn + 32, 0, 0);
        v16bf na0 = load_frag16(ap0 + kn);
        v16bf na1 = load_frag16(ap1 + kn);
        v16bf nb0 = load_frag16(bp + kn);
        v16bf nb1 = load_frag16(bp + 16 * EMB + kn);
        v16bf nb2 = load_frag16(bp + 32 * EMB + kn);
        v16bf nb3 = load_frag16(bp + 48 * EMB + kn);

        acc[0] = wmma_bf16(a0, b0, acc[0]);
        acc[1] = wmma_bf16(a0, b1, acc[1]);
        acc[2] = wmma_bf16(a0, b2, acc[2]);
        acc[3] = wmma_bf16(a0, b3, acc[3]);
        acc[4] = wmma_bf16(a1, b0, acc[4]);
        acc[5] = wmma_bf16(a1, b1, acc[5]);
        acc[6] = wmma_bf16(a1, b2, acc[6]);
        acc[7] = wmma_bf16(a1, b3, acc[7]);

        a0 = na0; a1 = na1;
        b0 = nb0; b1 = nb1; b2 = nb2; b3 = nb3;
    }

#pragma unroll
    for (int r = 0; r < 2; ++r) {
#pragma unroll
        for (int s = 0; s < 4; ++s) {
            const int n  = n0 + s * 16 + l15;
            const float bn = bias[n];
            const int h = n >> 6, d = n & 63;
#pragma unroll
            for (int j = 0; j < 8; ++j) {
                const int row = m0 + r * 16 + j + 8 * kh;
                const float val = acc[r * 4 + s][j] + bn;
                if (z == 2) {
                    Vt[((size_t)h * DH + d) * S_LEN + row] = (bf16_t)val;
                } else {
                    bf16_t* dst = (z == 0) ? Qb : Kb;
                    dst[((size_t)h * S_LEN + row) * DH + d] = (bf16_t)val;
                }
            }
        }
    }
}

// ---------------------------------------------------------------------------
// Kernel 4: flash attention per head, online softmax.
// One wave owns 16 query rows; streams keys in chunks of 32.  V fragments are
// issued before the softmax VALU chain so their latency hides under it.
// grid: (S/128, NH), block 256 (8 waves).
// ---------------------------------------------------------------------------
__global__ void attn_kernel(const bf16_t* __restrict__ Q,
                            const bf16_t* __restrict__ K,
                            const bf16_t* __restrict__ Vt,
                            bf16_t* __restrict__ AO) {
    __shared__ __align__(16) bf16_t pshare[8][16 * 32];

    const int h    = blockIdx.y;
    const int lane = threadIdx.x & 31;
    const int wave = threadIdx.x >> 5;
    const int kh   = lane >> 4;
    const int l15  = lane & 15;
    const int m0   = (blockIdx.x * 8 + wave) * 16;

    const bf16_t* Qh = Q  + (size_t)h * S_LEN * DH;
    const bf16_t* Kh = K  + (size_t)h * S_LEN * DH;
    const bf16_t* Vh = Vt + (size_t)h * DH * S_LEN;

    const bf16_t* qp = Qh + (size_t)(m0 + l15) * DH + kh * 8;
    v16bf aq0 = load_frag16(qp);
    v16bf aq1 = load_frag16(qp + 32);

    v8f o[4] = {};
    float rm[8], rl[8];
#pragma unroll
    for (int j = 0; j < 8; ++j) { rm[j] = -1e30f; rl[j] = 0.0f; }

    bf16_t* pl = &pshare[wave][0];
    const float sscale = 0.125f;   // 1/sqrt(64)

    for (int kv0 = 0; kv0 < S_LEN; kv0 += 32) {
        // ---- K fragments + score WMMAs
        const bf16_t* kb0 = Kh + (size_t)(kv0 + l15) * DH + kh * 8;
        const bf16_t* kb1 = Kh + (size_t)(kv0 + 16 + l15) * DH + kh * 8;
        v16bf bk00 = load_frag16(kb0);
        v16bf bk01 = load_frag16(kb0 + 32);
        v16bf bk10 = load_frag16(kb1);
        v16bf bk11 = load_frag16(kb1 + 32);

        // ---- V fragments issued EARLY (consumed after softmax)
        v16bf bv[4];
#pragma unroll
        for (int s4 = 0; s4 < 4; ++s4)
            bv[s4] = load_frag16(Vh + (size_t)(s4 * 16 + l15) * S_LEN + kv0 + kh * 8);

        v8f sc0 = {}, sc1 = {};
        sc0 = wmma_bf16(aq0, bk00, sc0);
        sc0 = wmma_bf16(aq1, bk01, sc0);
        sc1 = wmma_bf16(aq0, bk10, sc1);
        sc1 = wmma_bf16(aq1, bk11, sc1);

        // ---- online softmax update (row = j + 8*kh, replicated over 16 lanes)
#pragma unroll
        for (int j = 0; j < 8; ++j) {
            float s0 = sc0[j] * sscale;
            float s1 = sc1[j] * sscale;
            float mx = fmaxf(s0, s1);
#pragma unroll
            for (int off = 8; off >= 1; off >>= 1)
                mx = fmaxf(mx, __shfl_xor(mx, off, 32));
            const float mnew  = fmaxf(rm[j], mx);
            const float alpha = __expf(rm[j] - mnew);
            const float p0 = __expf(s0 - mnew);
            const float p1 = __expf(s1 - mnew);
            float rs = p0 + p1;
#pragma unroll
            for (int off = 8; off >= 1; off >>= 1)
                rs += __shfl_xor(rs, off, 32);
            rl[j] = rl[j] * alpha + rs;
            rm[j] = mnew;
#pragma unroll
            for (int s4 = 0; s4 < 4; ++s4) o[s4][j] *= alpha;

            const int prow = j + 8 * kh;
            pl[prow * 32 + l15]      = (bf16_t)p0;
            pl[prow * 32 + 16 + l15] = (bf16_t)p1;
        }
        asm volatile("s_wait_dscnt 0" ::: "memory");

        // ---- gather P as a 16x32 A-fragment (two ds_load_b128 per lane)
        v8bf plo = *(const v8bf*)&pshare[wave][l15 * 32 + kh * 8];
        v8bf phi = *(const v8bf*)&pshare[wave][l15 * 32 + kh * 8 + 16];
        v16bf ap = __builtin_shufflevector(plo, phi,
                                           0,1,2,3,4,5,6,7,8,9,10,11,12,13,14,15);

        // ---- O[16x64] += P[16x32] . V[32x64]
#pragma unroll
        for (int s4 = 0; s4 < 4; ++s4)
            o[s4] = wmma_bf16(ap, bv[s4], o[s4]);
    }

    // ---- normalize and store to AO[s][h*64+d]
#pragma unroll
    for (int s4 = 0; s4 < 4; ++s4) {
        const int d = s4 * 16 + l15;
#pragma unroll
        for (int j = 0; j < 8; ++j) {
            const int row = m0 + j + 8 * kh;
            AO[(size_t)row * EMB + h * DH + d] = (bf16_t)(o[s4][j] / rl[j]);
        }
    }
}

// ---------------------------------------------------------------------------
// Kernel 5: output projection. out[4096x768](f32) = AO(bf16) @ W_o + b_o
// Same 32x64-per-wave double-buffered structure. grid: (192), block 256.
// ---------------------------------------------------------------------------
__global__ void gemm_out_kernel(const bf16_t* __restrict__ AO,
                                const bf16_t* __restrict__ WtO,
                                const float* __restrict__ bo,
                                float* __restrict__ out) {
    const int lane = threadIdx.x & 31;
    const int wave = threadIdx.x >> 5;
    const int wgid = blockIdx.x * (blockDim.x >> 5) + wave;
    const int mt   = wgid / (EMB / 64);
    const int nt   = wgid % (EMB / 64);
    const int m0   = mt * 32;
    const int n0   = nt * 64;
    const int kh   = lane >> 4;
    const int l15  = lane & 15;

    const bf16_t* ap0 = AO  + (size_t)(m0 + l15) * EMB + kh * 8;
    const bf16_t* ap1 = AO  + (size_t)(m0 + 16 + l15) * EMB + kh * 8;
    const bf16_t* bp  = WtO + (size_t)(n0 + l15) * EMB + kh * 8;

    v8f acc[8] = {};
    v16bf a0 = load_frag16(ap0);
    v16bf a1 = load_frag16(ap1);
    v16bf b0 = load_frag16(bp);
    v16bf b1 = load_frag16(bp + 16 * EMB);
    v16bf b2 = load_frag16(bp + 32 * EMB);
    v16bf b3 = load_frag16(bp + 48 * EMB);

    for (int k0 = 0; k0 < EMB; k0 += 32) {
        int kn = k0 + 32;
        if (kn >= EMB) kn = 0;
        __builtin_prefetch(ap0 + kn + 32, 0, 0);
        v16bf na0 = load_frag16(ap0 + kn);
        v16bf na1 = load_frag16(ap1 + kn);
        v16bf nb0 = load_frag16(bp + kn);
        v16bf nb1 = load_frag16(bp + 16 * EMB + kn);
        v16bf nb2 = load_frag16(bp + 32 * EMB + kn);
        v16bf nb3 = load_frag16(bp + 48 * EMB + kn);

        acc[0] = wmma_bf16(a0, b0, acc[0]);
        acc[1] = wmma_bf16(a0, b1, acc[1]);
        acc[2] = wmma_bf16(a0, b2, acc[2]);
        acc[3] = wmma_bf16(a0, b3, acc[3]);
        acc[4] = wmma_bf16(a1, b0, acc[4]);
        acc[5] = wmma_bf16(a1, b1, acc[5]);
        acc[6] = wmma_bf16(a1, b2, acc[6]);
        acc[7] = wmma_bf16(a1, b3, acc[7]);

        a0 = na0; a1 = na1;
        b0 = nb0; b1 = nb1; b2 = nb2; b3 = nb3;
    }

#pragma unroll
    for (int r = 0; r < 2; ++r) {
#pragma unroll
        for (int s = 0; s < 4; ++s) {
            const int n = n0 + s * 16 + l15;
            const float bn = bo[n];
#pragma unroll
            for (int j = 0; j < 8; ++j) {
                const int row = m0 + r * 16 + j + 8 * kh;
                out[(size_t)row * EMB + n] = acc[r * 4 + s][j] + bn;
            }
        }
    }
}

// ---------------------------------------------------------------------------
extern "C" void kernel_launch(void* const* d_in, const int* in_sizes, int n_in,
                              void* d_out, int out_size, void* d_ws, size_t ws_size,
                              hipStream_t stream) {
    const float* q_in = (const float*)d_in[0];
    const float* k_in = (const float*)d_in[1];
    const float* v_in = (const float*)d_in[2];
    const float* W_q  = (const float*)d_in[3];
    const float* b_q  = (const float*)d_in[4];
    const float* W_k  = (const float*)d_in[5];
    const float* b_k  = (const float*)d_in[6];
    const float* W_v  = (const float*)d_in[7];
    const float* b_v  = (const float*)d_in[8];
    const float* W_o  = (const float*)d_in[9];
    const float* b_o  = (const float*)d_in[10];
    float* out = (float*)d_out;

    const size_t A   = (size_t)S_LEN * EMB;   // 3,145,728 elems
    const size_t WSZ = (size_t)EMB * EMB;     //   589,824 elems

    // workspace (bf16 elems): Xqkv[3A] | Wt[4W] | Q[A] | K[A] | Vt[A] | AO[A]
    bf16_t* Xq  = (bf16_t*)d_ws;
    bf16_t* Wt  = Xq + 3 * A;
    bf16_t* Qb  = Wt + 4 * WSZ;
    bf16_t* Kb  = Qb + A;
    bf16_t* Vtb = Kb + A;
    bf16_t* AO  = Vtb + A;

    convert_acts_kernel<<<dim3((unsigned)((3 * A / 4 + 255) / 256)), dim3(256), 0, stream>>>(
        q_in, k_in, v_in, Xq);

    transpose_w_kernel<<<dim3(EMB / 256, EMB, 4), dim3(256), 0, stream>>>(
        W_q, W_k, W_v, W_o, Wt);

    gemm_qkv_kernel<<<dim3(192, 1, 3), dim3(256), 0, stream>>>(
        Xq, Wt, b_q, b_k, b_v, Qb, Kb, Vtb);

    attn_kernel<<<dim3(S_LEN / 128, NH), dim3(256), 0, stream>>>(Qb, Kb, Vtb, AO);

    gemm_out_kernel<<<dim3(192), dim3(256), 0, stream>>>(AO, Wt + 3 * WSZ, b_o, out);
}
